// EfficientHybridCNN_85134841741329
// MI455X (gfx1250) — matrix operand order repaired and checked by reading, not compiled, not run on hardware
//
#include <hip/hip_runtime.h>
#include <hip/hip_bf16.h>
#include <math.h>

typedef __attribute__((ext_vector_type(16))) _Float16 v16h;
typedef __attribute__((ext_vector_type(8)))  _Float16 v8h;
typedef __attribute__((ext_vector_type(8)))  float    v8f;

#define APITCH 72   // 64 + 8 halves
#define BPITCH 72

// ---------------------------------------------------------------------------
// Fused conv + BN + ReLU as implicit GEMM on WMMA f32_16x16x32_f16.
//   GEMM: M = Cout (tile 32, blockIdx.y), N = B*OH*OW (tile 128, blockIdx.x),
//         K = Cin*KH*KW in stages of 64 (two WMMA k-steps), zero padded.
//   256 threads = 8 waves; wave wv owns N-subtile [wv*16, wv*16+16) and both
//   16-row M subtiles.  LDS double buffered: fill(stage+1) overlaps compute.
//   All geometry is compile-time -> divisions become magic multiplies.
// ---------------------------------------------------------------------------
template <int CIN, int H, int W, int COUT, int OH, int OW,
          int KH, int KW, int STRIDE, int PAD>
__global__ __launch_bounds__(256, 2)
void conv_bn_relu_wmma(const float* __restrict__ X, const float* __restrict__ Wt,
                       const float* __restrict__ cb,
                       const float* __restrict__ bnS, const float* __restrict__ bnB,
                       const float* __restrict__ bnM, const float* __restrict__ bnV,
                       float* __restrict__ Y)
{
    constexpr int KHW    = KH * KW;
    constexpr int KDIM   = CIN * KHW;
    constexpr int OHW    = OH * OW;
    constexpr int KSTEPS = (KDIM + 63) / 64;

    __shared__ __align__(16) _Float16 AsBuf[2 * 32 * APITCH];
    __shared__ __align__(16) _Float16 BsBuf[2 * 128 * BPITCH];

    const int tid  = threadIdx.x;
    const int lane = tid & 31;
    const int wv   = tid >> 5;               // wave id 0..7
    const int nBase = blockIdx.x * 128;
    const int mBase = blockIdx.y * 32;

    // ---- per-thread im2col column coordinates (fixed for whole K loop) ----
    const int colB  = tid & 127;             // n within tile
    const int khalf = tid >> 7;              // 0/1 -> which 32 K rows to fill
    const int nG   = nBase + colB;
    const int bB   = nG / OHW;               // compile-time OHW -> magic mul
    const int remB = nG - bB * OHW;
    const int oyB  = remB / OW;
    const int oxB  = remB - oyB * OW;
    const int iy0  = oyB * STRIDE - PAD;
    const int ix0  = oxB * STRIDE - PAD;

    // ---- stage filler: global -> LDS, branchless predicated loads ----
    auto fillStage = [&](int kStageBase, int bufIdx) {
        _Float16* As = AsBuf + bufIdx * (32 * APITCH);
        _Float16* Bs = BsBuf + bufIdx * (128 * BPITCH);

        // A (weights): 32x64 halves, 8 contiguous per thread, one b128 store.
        {
            const int e  = tid * 8;
            const int m  = e >> 6;
            const int k0 = e & 63;
            float tw[8];
            #pragma unroll
            for (int u = 0; u < 8; ++u) {
                const int  kg = kStageBase + k0 + u;
                const bool p  = (kg < KDIM);
                tw[u] = Wt[p ? ((mBase + m) * KDIM + kg) : 0];
                tw[u] = p ? tw[u] : 0.f;
            }
            v8h av;
            #pragma unroll
            for (int u = 0; u < 8; ++u) av[u] = (_Float16)tw[u];
            *(v8h*)(As + m * APITCH + k0) = av;
        }
        // B (im2col): 128 cols x 64 K; this thread: col colB, 32 K values,
        // in 4 groups of 8 (clause of 8 loads -> 1 wait -> 1 b128 store).
        #pragma unroll
        for (int g = 0; g < 32; g += 8) {
            float tv[8];
            #pragma unroll
            for (int u = 0; u < 8; ++u) {
                const int k  = kStageBase + khalf * 32 + g + u;
                bool p = (k < KDIM);
                const int kc = p ? k : 0;
                const int c  = kc / KHW;           // compile-time consts
                const int r2 = kc - c * KHW;
                const int ky = r2 / KW;
                const int kx = r2 - ky * KW;
                const int iy = iy0 + ky;
                const int ix = ix0 + kx;
                p = p && ((unsigned)iy < (unsigned)H) && ((unsigned)ix < (unsigned)W);
                const long a = p ? ((long)(bB * CIN + c) * H + iy) * W + ix : 0;
                tv[u] = X[a];
                tv[u] = p ? tv[u] : 0.f;
            }
            v8h bv;
            #pragma unroll
            for (int u = 0; u < 8; ++u) bv[u] = (_Float16)tv[u];
            *(v8h*)(Bs + colB * BPITCH + khalf * 32 + g) = bv;
        }
    };

    v8f acc0 = {};
    v8f acc1 = {};
    const int rowA0 = (lane & 15);
    const int rowA1 = 16 + (lane & 15);
    const int rB    = wv * 16 + (lane & 15);

    fillStage(0, 0);
    __syncthreads();

    for (int ks = 0; ks < KSTEPS; ++ks) {
        if (ks + 1 < KSTEPS) fillStage((ks + 1) * 64, (ks + 1) & 1);

        const _Float16* As = AsBuf + (ks & 1) * (32 * APITCH);
        const _Float16* Bs = BsBuf + (ks & 1) * (128 * BPITCH);

        #pragma unroll
        for (int s = 0; s < 2; ++s) {            // two K=32 WMMA steps
            const int kOff  = s * 32;
            const int kOffB = kOff + ((lane >> 4) << 4);
            v16h bf;
            #pragma unroll
            for (int j = 0; j < 16; ++j) bf[j] = Bs[rB * BPITCH + kOffB + j];

            const int kHiA = kOff + ((lane >> 4) << 4);
            v16h af0, af1;
            #pragma unroll
            for (int j = 0; j < 8; ++j) {
                af0[j]     = As[rowA0 * APITCH + kHiA + j];
                af0[j + 8] = As[rowA0 * APITCH + kHiA + 8 + j];
                af1[j]     = As[rowA1 * APITCH + kHiA + j];
                af1[j + 8] = As[rowA1 * APITCH + kHiA + 8 + j];
            }
            acc0 = __builtin_amdgcn_wmma_f32_16x16x32_f16(false, af0, false, bf,
                                                          (short)0, acc0, false, false);
            acc1 = __builtin_amdgcn_wmma_f32_16x16x32_f16(false, af1, false, bf,
                                                          (short)0, acc1, false, false);
        }
        __syncthreads();
    }

    // ---- epilogue: fused bias + BN + ReLU; lanes 0..15 of each half-wave
    // write 16 consecutive floats per channel row (coalesced). ----
    const int nOut = nBase + wv * 16 + (lane & 15);
    const int bO   = nOut / OHW;
    const int remO = nOut - bO * OHW;
    const int oy   = remO / OW;
    const int ox   = remO - oy * OW;
    #pragma unroll
    for (int mt = 0; mt < 2; ++mt) {
        const v8f acc = mt ? acc1 : acc0;
        #pragma unroll
        for (int r = 0; r < 8; ++r) {
            const int   mOut = mBase + mt * 16 + ((lane >> 4) << 3) + r;
            const float scl  = bnS[mOut] * rsqrtf(bnV[mOut] + 1e-5f);
            const float bia  = (cb[mOut] - bnM[mOut]) * scl + bnB[mOut];
            const float y    = fmaxf(fmaf(acc[r], scl, bia), 0.f);
            Y[((bO * COUT + mOut) * OH + oy) * OW + ox] = y;
        }
    }
}

// ---------------------------------------------------------------------------
// AdaptiveAvgPool2d((2,2)) on 14x14 == exact 7x7 block mean.  (B,128,14,14)
// -> (B,512) with feature index c*4 + py*2 + px.
// ---------------------------------------------------------------------------
__global__ void pool_kernel(const float* __restrict__ h3, float* __restrict__ pooled,
                            int total)
{
    const int o = blockIdx.x * blockDim.x + threadIdx.x;
    if (o >= total) return;
    const int b   = o >> 9;
    const int rem = o & 511;
    const int c   = rem >> 2;
    const int p   = rem & 3;
    const int py  = p >> 1;
    const int px  = p & 1;
    const float* base = h3 + ((b * 128 + c) * 14 + py * 7) * 14 + px * 7;
    float s = 0.f;
    #pragma unroll
    for (int yy = 0; yy < 7; ++yy)
        #pragma unroll
        for (int xx = 0; xx < 7; ++xx)
            s += base[yy * 14 + xx];
    pooled[o] = s * (1.f / 49.f);
}

// ---------------------------------------------------------------------------
// 4-qubit statevector gates on 16 complex amplitudes.
// Flat index bit for wire w is (3-w) (jnp axis-0 is most significant).
// ---------------------------------------------------------------------------
__device__ __forceinline__ void q_ry(float* sr, float* si, int w, float t) {
    const float c = cosf(t * 0.5f), s = sinf(t * 0.5f);
    const int st = 1 << (3 - w);
    #pragma unroll
    for (int i = 0; i < 16; ++i) {
        if (i & st) continue;
        const int j = i | st;
        const float a0r = sr[i], a0i = si[i], a1r = sr[j], a1i = si[j];
        sr[i] = c * a0r - s * a1r;  si[i] = c * a0i - s * a1i;
        sr[j] = s * a0r + c * a1r;  si[j] = s * a0i + c * a1i;
    }
}
__device__ __forceinline__ void q_rx(float* sr, float* si, int w, float t) {
    const float c = cosf(t * 0.5f), s = sinf(t * 0.5f);
    const int st = 1 << (3 - w);
    #pragma unroll
    for (int i = 0; i < 16; ++i) {
        if (i & st) continue;
        const int j = i | st;
        const float a0r = sr[i], a0i = si[i], a1r = sr[j], a1i = si[j];
        sr[i] = c * a0r + s * a1i;   si[i] = c * a0i - s * a1r;
        sr[j] = s * a0i + c * a1r;   si[j] = -s * a0r + c * a1i;
    }
}
__device__ __forceinline__ void q_rz(float* sr, float* si, int w, float t) {
    const float c = cosf(t * 0.5f), s = sinf(t * 0.5f);
    const int st = 1 << (3 - w);
    #pragma unroll
    for (int i = 0; i < 16; ++i) {
        const float r = sr[i], im = si[i];
        if (i & st) { sr[i] = c * r - s * im;  si[i] = c * im + s * r; }   // e^{+it/2}
        else        { sr[i] = c * r + s * im;  si[i] = c * im - s * r; }   // e^{-it/2}
    }
}
__device__ __forceinline__ void q_cnot(float* sr, float* si, int c, int t) {
    const int sc = 1 << (3 - c), stt = 1 << (3 - t);
    #pragma unroll
    for (int i = 0; i < 16; ++i) {
        if ((i & sc) && !(i & stt)) {
            const int j = i | stt;
            float tr = sr[i]; sr[i] = sr[j]; sr[j] = tr;
            float ti = si[i]; si[i] = si[j]; si[j] = ti;
        }
    }
}

// ---------------------------------------------------------------------------
// Head: pre-linear + tanh*pi -> quantum circuit -> post1(relu) -> post2.
// One thread per sample (work is tiny: <5 KFLOP/sample).
// ---------------------------------------------------------------------------
__global__ void head_kernel(const float* __restrict__ pooled,
                            const float* __restrict__ preW, const float* __restrict__ preB,
                            const float* __restrict__ qW,
                            const float* __restrict__ p1W, const float* __restrict__ p1B,
                            const float* __restrict__ p2W, const float* __restrict__ p2B,
                            float* __restrict__ out, int B)
{
    const int i = blockIdx.x * blockDim.x + threadIdx.x;
    if (i >= B) return;
    const float* h = pooled + i * 512;

    float ang[4];
    #pragma unroll
    for (int q = 0; q < 4; ++q) {
        float a = preB[q];
        const float* wrow = preW + q * 512;
        for (int j = 0; j < 512; ++j) a = fmaf(h[j], wrow[j], a);
        ang[q] = tanhf(a) * 3.14159265358979323846f;
    }

    float sr[16], si[16];
    #pragma unroll
    for (int k = 0; k < 16; ++k) { sr[k] = 0.f; si[k] = 0.f; }
    sr[0] = 1.f;

    #pragma unroll
    for (int q = 0; q < 4; ++q) q_ry(sr, si, q, ang[q]);
    #pragma unroll
    for (int l = 0; l < 2; ++l) {
        #pragma unroll
        for (int q = 0; q < 4; ++q) {
            const float* wg = qW + (l * 4 + q) * 3;
            q_rx(sr, si, q, wg[0]);
            q_ry(sr, si, q, wg[1]);
            q_rz(sr, si, q, wg[2]);
        }
        #pragma unroll
        for (int q = 0; q < 3; ++q) q_cnot(sr, si, q, q + 1);
    }

    float ev[4];
    #pragma unroll
    for (int q = 0; q < 4; ++q) {
        const int mask = 1 << (3 - q);
        float e = 0.f;
        #pragma unroll
        for (int k = 0; k < 16; ++k) {
            const float p = sr[k] * sr[k] + si[k] * si[k];
            e += (k & mask) ? -p : p;
        }
        ev[q] = e;
    }

    float o1[64];
    #pragma unroll
    for (int j = 0; j < 64; ++j) {
        float a = p1B[j];
        #pragma unroll
        for (int q = 0; q < 4; ++q) a = fmaf(ev[q], p1W[j * 4 + q], a);
        o1[j] = fmaxf(a, 0.f);
    }
    #pragma unroll
    for (int k = 0; k < 5; ++k) {
        float a = p2B[k];
        for (int j = 0; j < 64; ++j) a = fmaf(o1[j], p2W[k * 64 + j], a);
        out[i * 5 + k] = a;
    }
}

// ---------------------------------------------------------------------------
extern "C" void kernel_launch(void* const* d_in, const int* in_sizes, int n_in,
                              void* d_out, int out_size, void* d_ws, size_t ws_size,
                              hipStream_t stream)
{
    const float* x    = (const float*)d_in[0];
    const float* c1w  = (const float*)d_in[1];
    const float* c1b  = (const float*)d_in[2];
    const float* bn1s = (const float*)d_in[3];
    const float* bn1b = (const float*)d_in[4];
    const float* bn1m = (const float*)d_in[5];
    const float* bn1v = (const float*)d_in[6];
    const float* c2w  = (const float*)d_in[7];
    const float* c2b  = (const float*)d_in[8];
    const float* bn2s = (const float*)d_in[9];
    const float* bn2b = (const float*)d_in[10];
    const float* bn2m = (const float*)d_in[11];
    const float* bn2v = (const float*)d_in[12];
    const float* c3w  = (const float*)d_in[13];
    const float* c3b  = (const float*)d_in[14];
    const float* bn3s = (const float*)d_in[15];
    const float* bn3b = (const float*)d_in[16];
    const float* bn3m = (const float*)d_in[17];
    const float* bn3v = (const float*)d_in[18];
    const float* preW = (const float*)d_in[19];
    const float* preB = (const float*)d_in[20];
    const float* qW   = (const float*)d_in[21];
    const float* p1W  = (const float*)d_in[22];
    const float* p1B  = (const float*)d_in[23];
    const float* p2W  = (const float*)d_in[24];
    const float* p2B  = (const float*)d_in[25];
    float* out = (float*)d_out;

    const int B = 256;

    // Workspace carve-out (floats): h1 25.69M, h2 12.85M, h3 6.42M, pooled 128K.
    float* h1     = (float*)d_ws;
    float* h2     = h1 + (size_t)B * 32 * 56 * 56;
    float* h3     = h2 + (size_t)B * 64 * 28 * 28;
    float* pooled = h3 + (size_t)B * 128 * 14 * 14;

    // conv1: 3->32, 7x7 s4 p3, 224 -> 56.  N = 802816 = 128*6272.
    conv_bn_relu_wmma<3, 224, 224, 32, 56, 56, 7, 7, 4, 3>
        <<<dim3(6272, 1), 256, 0, stream>>>(x, c1w, c1b, bn1s, bn1b, bn1m, bn1v, h1);

    // conv2: 32->64, 3x3 s2 p1, 56 -> 28.  N = 200704 = 128*1568, M blocks = 2.
    conv_bn_relu_wmma<32, 56, 56, 64, 28, 28, 3, 3, 2, 1>
        <<<dim3(1568, 2), 256, 0, stream>>>(h1, c2w, c2b, bn2s, bn2b, bn2m, bn2v, h2);

    // conv3: 64->128, 3x3 s2 p1, 28 -> 14.  N = 50176 = 128*392, M blocks = 4.
    conv_bn_relu_wmma<64, 28, 28, 128, 14, 14, 3, 3, 2, 1>
        <<<dim3(392, 4), 256, 0, stream>>>(h2, c3w, c3b, bn3s, bn3b, bn3m, bn3v, h3);

    // pool -> (B,512)
    const int poolTotal = B * 512;
    pool_kernel<<<(poolTotal + 255) / 256, 256, 0, stream>>>(h3, pooled, poolTotal);

    // head -> (B,5)
    head_kernel<<<(B + 255) / 256, 256, 0, stream>>>(
        pooled, preW, preB, qW, p1W, p1B, p2W, p2B, out, B);
}
